// Stack_59081570124278
// MI455X (gfx1250) — compile-verified, matrix operand order checked by reading, not run on hardware
//
#include <hip/hip_runtime.h>

typedef __attribute__((ext_vector_type(16))) _Float16 v16h;
typedef __attribute__((ext_vector_type(8)))  float    v8f;

#define TNUM 512
#define BNUM 128
#define ENUM 256
#define TT   32      // timestep tile accumulated in LDS before each WMMA flush
#define RPAD 6       // 6 f16 = 3 dwords row pad -> conflict-free fragment reads

__global__ __launch_bounds__(512)
void stack_fused_kernel(const float* __restrict__ v,
                        const float* __restrict__ u,
                        const float* __restrict__ d,
                        float* __restrict__ out)
{
    const int b    = blockIdx.x;          // one block per batch
    const int tid  = threadIdx.x;         // thread owns stack slot i = tid
    const int wv   = tid >> 5;            // wave id 0..15
    const int lane = tid & 31;
    const int half = lane >> 4;           // 0|1 (lane group)
    const int l16  = lane & 15;

    __shared__ float    waveSums[2][16];           // parity double-buffered scan partials
    __shared__ _Float16 Ctile[TT][TNUM + RPAD];    // coeff tile, rows = t (mod TT)
    __shared__ _Float16 Vlds[ENUM][TNUM + RPAD];   // V_b transposed: [e][k], f16

    const float* vB = v + b * ENUM;       // V_b(k,e) at vB[k*B*E + e]

    // ---- stage V_b into LDS as f16 (transposed so B K-pairs are dword-contiguous)
    for (int idx = tid; idx < TNUM * ENUM; idx += 512) {
        const int k = idx >> 8;           // 0..511
        const int e = idx & 255;          // 0..255 (consecutive across lanes -> coalesced)
        Vlds[e][k] = (_Float16)vB[k * (BNUM * ENUM) + e];
    }
    __syncthreads();

    float S = 0.f;                        // this slot's strength
    float P = 0.f, total = 0.f;           // inclusive prefix of S, and its total (current S)

    for (int t = 0; t < TNUM; ++t) {
        const float u_t = u[t * BNUM + b];
        const float d_t = d[t * BNUM + b];

        // ---- pop phase: exclusive suffix sum = total - inclusive prefix (from last scan)
        float R = total - P;
        float w = fmaxf(u_t - R, 0.f);
        S = fmaxf(S - w, 0.f);
        if (tid == t) S = d_t;            // push

        // ---- single block-wide inclusive prefix scan of updated S
        float p = S;
        #pragma unroll
        for (int o = 1; o < 32; o <<= 1) {
            float n = __shfl_up(p, o, 32);
            if (lane >= o) p += n;
        }
        if (lane == 31) waveSums[t & 1][wv] = p;
        __syncthreads();
        float off = 0.f, tot = 0.f;
        #pragma unroll
        for (int ww = 0; ww < 16; ++ww) {
            float s = waveSums[t & 1][ww];
            tot += s;
            if (ww < wv) off += s;
        }
        P = p + off;
        total = tot;

        // ---- read coefficients for this step; stash row in LDS tile (f16)
        float c = fminf(S, fmaxf(1.f - (total - P), 0.f));
        Ctile[t & (TT - 1)][tid] = (_Float16)c;

        // ---- every TT steps: reads[t0..t0+TT-1, b, :] = Ctile(TTx512) @ V_b(512x256)
        if ((t & (TT - 1)) == (TT - 1)) {
            __syncthreads();              // Ctile rows complete
            const int t0 = t - (TT - 1);
            const int e0 = wv * 16;       // 16 waves tile E = 256
            v8f acc0 = {};
            v8f acc1 = {};
            for (int k0 = 0; k0 < TNUM; k0 += 32) {
                // B fragment from LDS: V_b[k0..k0+31, e0..e0+15] (lane = N, K pairs/VGPR)
                v16h bf;
                #pragma unroll
                for (int j = 0; j < 8; ++j) {
                    const int kk = k0 + ((j < 4) ? 2 * j : 2 * j + 8) + 8 * half;
                    bf[2 * j]     = Vlds[e0 + l16][kk];
                    bf[2 * j + 1] = Vlds[e0 + l16][kk + 1];
                }
                // A fragments from LDS: Ctile rows (rt*16 + l16), K pairs per VGPR
                v16h a0, a1;
                #pragma unroll
                for (int j = 0; j < 8; ++j) {
                    const int kk = k0 + ((j < 4) ? 2 * j : 2 * j + 8) + 8 * half;
                    a0[2 * j]     = Ctile[l16][kk];
                    a0[2 * j + 1] = Ctile[l16][kk + 1];
                    a1[2 * j]     = Ctile[16 + l16][kk];
                    a1[2 * j + 1] = Ctile[16 + l16][kk + 1];
                }
                acc0 = __builtin_amdgcn_wmma_f32_16x16x32_f16(
                    false, a0, false, bf, (short)0, acc0, false, false);
                acc1 = __builtin_amdgcn_wmma_f32_16x16x32_f16(
                    false, a1, false, bf, (short)0, acc1, false, false);
            }
            // D store: VGPR r -> row r + 8*half, col = e0 + l16
            #pragma unroll
            for (int r = 0; r < 8; ++r) {
                const int m = r + 8 * half;
                out[(t0 + m)      * (BNUM * ENUM) + b * ENUM + e0 + l16] = acc0[r];
                out[(t0 + 16 + m) * (BNUM * ENUM) + b * ENUM + e0 + l16] = acc1[r];
            }
            __syncthreads();              // done reading Ctile before it is overwritten
        }
    }
}

extern "C" void kernel_launch(void* const* d_in, const int* in_sizes, int n_in,
                              void* d_out, int out_size, void* d_ws, size_t ws_size,
                              hipStream_t stream) {
    (void)in_sizes; (void)n_in; (void)d_ws; (void)ws_size; (void)out_size;
    const float* v = (const float*)d_in[0];   // [T, B, E]
    const float* u = (const float*)d_in[1];   // [T, B]
    const float* d = (const float*)d_in[2];   // [T, B]
    float* out = (float*)d_out;               // [T, B, E]
    stack_fused_kernel<<<dim3(BNUM), dim3(512), 0, stream>>>(v, u, d, out);
}
